// Spatial_Temporal_Attention_652835029262
// MI455X (gfx1250) — matrix operand order repaired and checked by reading, not compiled, not run on hardware
//
#include <hip/hip_runtime.h>

// ---------------------------------------------------------------------------
// Spatial-Temporal attention for MI455X (gfx1250, wave32, WMMA).
// - All GEMMs: v_wmma_f32_16x16x32_bf16, f32 accumulation.
// - B operands live in GLOBAL memory pre-swizzled in fragment-major 32x16
//   tiles, so LDS staging is a pure linear copy and every WMMA operand load
//   is a contiguous ds_load_b128 pair.
// - Staging uses global_load_async_to_lds_b128 with double-buffered LDS
//   (ASYNCcnt pipelining); falls back to sync copies if builtins absent.
// ---------------------------------------------------------------------------

typedef __attribute__((ext_vector_type(16))) __bf16 v16bf;
typedef __attribute__((ext_vector_type(8)))  __bf16 bf8v;
typedef __attribute__((ext_vector_type(8)))  float  v8f;
typedef __attribute__((ext_vector_type(4)))  int    v4i;

#if defined(__has_builtin)
#if __has_builtin(__builtin_amdgcn_global_load_async_to_lds_b128) && \
    __has_builtin(__builtin_amdgcn_s_wait_asynccnt)
#define HAVE_ASYNC 1
#endif
#endif
#ifndef HAVE_ASYNC
#define HAVE_ASYNC 0
#endif

#define B_   16
#define T_   1024
#define N_   512
#define DKT  64
#define DKS  128

#define BM 128
#define BN 64
#define BK 32
#define GEMM_THREADS 128   // 4 waves; each wave owns 32 rows x 64 cols

#define LDA_S (BK + 8)     // 40 halves = 80 B rows (16B-divisible)

// --- global fragment-major swizzle for B operands ---------------------------
// B[k][n] (k in K-dim, n in N-dim, ncols = total N, multiple of 16; K mult 32)
// tile = (k/32, n/16) -> 512 contiguous halves laid out [lane][i]:
//   lane = (n%16) | (k&16), i = k%16
__device__ inline long long bswz(int k, int n, int ncols) {
  return ((long long)(k >> 5) * (ncols >> 4) + (n >> 4)) * 512
       + (long long)((((n & 15) | (k & 16)) << 4) + (k & 15));
}

// --- WMMA A fragment (ISA 7.12.2): 16x32 bf16 from row-major LDS -----------
__device__ inline v16bf frag_a(const __bf16* base, int lane) {
  const int row = lane & 15;
  const int kg  = (lane >> 4) << 4;
  const __bf16* p = base + row * LDA_S + kg;
  bf8v lo = *(const bf8v*)(p);
  bf8v hi = *(const bf8v*)(p + 8);
  return __builtin_shufflevector(lo, hi, 0, 1, 2, 3, 4, 5, 6, 7,
                                         8, 9, 10, 11, 12, 13, 14, 15);
}

#if HAVE_ASYNC
typedef __attribute__((address_space(1))) v4i gv4i;   // global int4
typedef __attribute__((address_space(3))) v4i lv4i;   // LDS int4

__device__ inline void async_b128(const __bf16* g, __bf16* l) {
  __builtin_amdgcn_global_load_async_to_lds_b128((gv4i*)(g), (lv4i*)(l), 0, 0);
}
#endif

__device__ inline void stage_sync(const __bf16* __restrict__ Ab, int lda, int k0,
                                  const __bf16* __restrict__ Bchunk,
                                  __bf16 (*lA)[LDA_S], __bf16* lB, int tid) {
  const uint4* ga = reinterpret_cast<const uint4*>(Ab + (long long)tid * lda + k0);
  uint4* sa = reinterpret_cast<uint4*>(&lA[tid][0]);
  sa[0] = ga[0]; sa[1] = ga[1]; sa[2] = ga[2]; sa[3] = ga[3];
  const uint4* gb = reinterpret_cast<const uint4*>(Bchunk) + tid * 2;
  uint4* sb = reinterpret_cast<uint4*>(lB) + tid * 2;
  sb[0] = gb[0]; sb[1] = gb[1];
}

#if HAVE_ASYNC
// 6 async b128 wave-ops per wave per tile: 4 for A (8 KB), 2 for B (4 KB)
__device__ inline void stage_async(const __bf16* __restrict__ Ab, int lda, int k0,
                                   const __bf16* __restrict__ Bchunk,
                                   __bf16 (*lA)[LDA_S], __bf16* lB, int tid) {
#pragma unroll
  for (int i = 0; i < 4; ++i) {
    const int chunk = tid + 128 * i;            // 512 x 16B chunks of A tile
    const int row = chunk >> 2, coff = (chunk & 3) * 8;
    async_b128(Ab + (long long)row * lda + k0 + coff, &lA[row][coff]);
  }
#pragma unroll
  for (int i = 0; i < 2; ++i) {
    const int e = (tid + 128 * i) * 8;          // linear 4 KB copy
    async_b128(Bchunk + e, lB + e);
  }
}
#endif

__device__ inline void compute_tile(const __bf16 (*lA)[LDA_S], const __bf16* lB,
                                    int lane, int wave, v8f acc[2][4]) {
  const v16bf a0 = frag_a(&lA[wave * 32][0], lane);
  const v16bf a1 = frag_a(&lA[wave * 32 + 16][0], lane);
#pragma unroll
  for (int j = 0; j < 4; ++j) {
    const v16bf bfr = *(const v16bf*)(lB + j * 512 + lane * 16);
    acc[0][j] = __builtin_amdgcn_wmma_f32_16x16x32_bf16(
        false, a0, false, bfr, (short)0, acc[0][j], false, false);
    acc[1][j] = __builtin_amdgcn_wmma_f32_16x16x32_bf16(
        false, a1, false, bfr, (short)0, acc[1][j], false, false);
  }
}

// --- Universal bf16 GEMM, C = A(MxK) * B_swz(KxN), f32 accumulate -----------
// mode 0: f32 row-major                       (logits)
// mode 1: bf16 row-major                      (outT, out2 -> A consumers)
// mode 2: bf16 swizzled-B of C^T              (K projections -> B consumers)
// mode 3: bf16 row-major, v*scale+bias[col]   (Q projections w/ logit bias)
// mode 4: f32 transposed, v+bias[col]         (final + bout -> d_out[b,n,t])
// mode 5: bf16 swizzled-B of C                (V -> B consumer)
__global__ __launch_bounds__(GEMM_THREADS)
void gemm_bf16_kernel(const __bf16* __restrict__ A, long long strideA, int lda,
                      const __bf16* __restrict__ Bm, long long strideB, int ldb,
                      float* __restrict__ Cf, __bf16* __restrict__ Cb,
                      long long strideC, int ldc,
                      const float* __restrict__ bias, float scale,
                      int K, int mode) {
#if HAVE_ASYNC
  __shared__ __attribute__((aligned(32))) __bf16 lA[2][BM][LDA_S];   // 20 KB
  __shared__ __attribute__((aligned(32))) __bf16 lB[2][4 * 512];     //  8 KB
#else
  __shared__ __attribute__((aligned(32))) __bf16 lA[1][BM][LDA_S];
  __shared__ __attribute__((aligned(32))) __bf16 lB[1][4 * 512];
#endif

  const int tid  = threadIdx.x;
  const int lane = tid & 31;
  const int wave = tid >> 5;
  const int bz   = blockIdx.z;

  const __bf16* Ab = A + (long long)bz * strideA + (long long)(blockIdx.y * BM) * lda;
  const long long kChunk = (long long)(ldb >> 4) * 512;   // halves per k-tile row
  const __bf16* Bb = Bm + (long long)bz * strideB + (long long)blockIdx.x * 4 * 512;

  v8f acc[2][4];
#pragma unroll
  for (int a = 0; a < 2; ++a)
#pragma unroll
    for (int j = 0; j < 4; ++j)
#pragma unroll
      for (int r = 0; r < 8; ++r) acc[a][j][r] = 0.0f;

  const int nsteps = K / BK;
#if HAVE_ASYNC
  stage_async(Ab, lda, 0, Bb, lA[0], lB[0], tid);
  for (int s = 0; s < nsteps; ++s) {
    const int cur = s & 1;
    if (s + 1 < nsteps) {
      stage_async(Ab, lda, (s + 1) * BK, Bb + (long long)(s + 1) * kChunk,
                  lA[cur ^ 1], lB[cur ^ 1], tid);
      __builtin_amdgcn_s_wait_asynccnt(6);   // tile s complete (6 newer pending)
    } else {
      __builtin_amdgcn_s_wait_asynccnt(0);
    }
    __syncthreads();                          // all waves' tile-s data visible
    compute_tile(lA[cur], lB[cur], lane, wave, acc);
    __syncthreads();                          // release buffer before overwrite
  }
#else
  for (int s = 0; s < nsteps; ++s) {
    __syncthreads();
    stage_sync(Ab, lda, s * BK, Bb + (long long)s * kChunk, lA[0], lB[0], tid);
    __syncthreads();
    compute_tile(lA[0], lB[0], lane, wave, acc);
  }
#endif

  // C/D layout: lane 0-15 -> N=lane, VGPR r -> M=r (lanes 0-15) / M=8+r
  const int colLane = lane & 15;
  const long long cbase = (long long)bz * strideC;
#pragma unroll
  for (int a = 0; a < 2; ++a) {
    const int rowBase = blockIdx.y * BM + wave * 32 + a * 16 + ((lane >> 4) << 3);
#pragma unroll
    for (int j = 0; j < 4; ++j) {
      const int col = blockIdx.x * BN + j * 16 + colLane;
#pragma unroll
      for (int r = 0; r < 8; ++r) {
        const int row = rowBase + r;
        const float v = acc[a][j][r];
        if (mode == 0)      Cf[cbase + (long long)row * ldc + col] = v;
        else if (mode == 1) Cb[cbase + (long long)row * ldc + col] = (__bf16)v;
        else if (mode == 2) Cb[cbase + bswz(col, row, ldc)] = (__bf16)v;
        else if (mode == 3) Cb[cbase + (long long)row * ldc + col] =
                                (__bf16)(v * scale + bias[col]);
        else if (mode == 4) Cf[cbase + (long long)col * ldc + row] = v + bias[col];
        else                Cb[cbase + bswz(row, col, ldc)] = (__bf16)v;
      }
    }
  }
}

// --- Row softmax: f32 logits -> bf16 probs; optional swizzled-B^T store -----
__global__ __launch_bounds__(256)
void softmax_bf16_kernel(const float* __restrict__ src, __bf16* __restrict__ dst,
                         int cols, int rowsPerBatch, long long dstBatchStride,
                         int ldd, int swizzleT) {
  const int gr = blockIdx.x;
  const int b  = gr / rowsPerBatch;
  const int r  = gr - b * rowsPerBatch;
  const float* s = src + (long long)gr * cols;
  __bf16* d = dst + (long long)b * dstBatchStride;

  __shared__ float red[256];
  const int tid = threadIdx.x;

  float mx = -3.4e38f;
  for (int c = tid; c < cols; c += 256) mx = fmaxf(mx, s[c]);
  red[tid] = mx; __syncthreads();
  for (int off = 128; off > 0; off >>= 1) {
    if (tid < off) red[tid] = fmaxf(red[tid], red[tid + off]);
    __syncthreads();
  }
  mx = red[0]; __syncthreads();

  float sum = 0.0f;
  for (int c = tid; c < cols; c += 256) sum += __expf(s[c] - mx);
  red[tid] = sum; __syncthreads();
  for (int off = 128; off > 0; off >>= 1) {
    if (tid < off) red[tid] += red[tid + off];
    __syncthreads();
  }
  const float inv = 1.0f / red[0];

  for (int c = tid; c < cols; c += 256) {
    const float v = __expf(s[c] - mx) * inv;
    if (swizzleT) d[bswz(c, r, ldd)] = (__bf16)v;   // attn^T, fragment-major
    else          d[(long long)r * ldd + c] = (__bf16)v;
  }
}

__global__ void f32_to_bf16_kernel(const float* __restrict__ s,
                                   __bf16* __restrict__ d, long long n) {
  long long i = (long long)blockIdx.x * blockDim.x + threadIdx.x;
  const long long stride = (long long)gridDim.x * blockDim.x;
  for (; i < n; i += stride) d[i] = (__bf16)s[i];
}

__global__ void f32_to_bf16_swz_kernel(const float* __restrict__ s,
                                       __bf16* __restrict__ d,
                                       int rows, int cols) {
  const long long n = (long long)rows * cols;
  long long i = (long long)blockIdx.x * blockDim.x + threadIdx.x;
  const long long stride = (long long)gridDim.x * blockDim.x;
  for (; i < n; i += stride) {
    const int k = (int)(i / cols), c = (int)(i - (long long)k * cols);
    d[bswz(k, c, cols)] = (__bf16)s[i];
  }
}

// ---------------------------------------------------------------------------
extern "C" void kernel_launch(void* const* d_in, const int* in_sizes, int n_in,
                              void* d_out, int out_size, void* d_ws, size_t ws_size,
                              hipStream_t stream) {
  const float* x_T    = (const float*)d_in[0];
  const float* x_S    = (const float*)d_in[1];
  const float* Wq_T   = (const float*)d_in[2];
  const float* Wk_T   = (const float*)d_in[3];
  const float* Wv_T   = (const float*)d_in[4];
  const float* Wq_S   = (const float*)d_in[5];
  const float* Wk_S   = (const float*)d_in[6];
  const float* Wout_T = (const float*)d_in[7];
  const float* bout_T = (const float*)d_in[8];
  const float* bias_T = (const float*)d_in[9];
  const float* bias_S = (const float*)d_in[10];
  float* out = (float*)d_out;

  char* ws = (char*)d_ws;
  size_t off = 0;
  auto alloc = [&](long long elems, int esize) -> void* {
    void* p = ws + off;
    off = (off + (size_t)elems * esize + 255) & ~(size_t)255;
    return p;
  };

  __bf16* xT_bf  = (__bf16*)alloc((long long)B_ * T_ * N_, 2);
  __bf16* xS_bf  = (__bf16*)alloc((long long)B_ * N_ * T_, 2);
  __bf16* wqT    = (__bf16*)alloc((long long)N_ * DKT, 2);   // swizzled
  __bf16* wkT    = (__bf16*)alloc((long long)N_ * DKT, 2);   // swizzled
  __bf16* wvT    = (__bf16*)alloc((long long)N_ * N_, 2);    // swizzled
  __bf16* wqS    = (__bf16*)alloc((long long)T_ * DKS, 2);   // swizzled
  __bf16* wkS    = (__bf16*)alloc((long long)T_ * DKS, 2);   // swizzled
  __bf16* wout   = (__bf16*)alloc((long long)N_ * N_, 2);    // swizzled
  __bf16* qT     = (__bf16*)alloc((long long)B_ * T_ * DKT, 2);
  __bf16* kTt    = (__bf16*)alloc((long long)B_ * DKT * T_, 2);  // swizzled
  __bf16* vT     = (__bf16*)alloc((long long)B_ * T_ * N_, 2);   // swizzled
  float*  logits = (float*) alloc((long long)B_ * T_ * T_, 4);   // reused for S
  __bf16* attnT  = (__bf16*)alloc((long long)B_ * T_ * T_, 2);
  __bf16* outT   = (__bf16*)alloc((long long)B_ * T_ * N_, 2);
  __bf16* qS     = (__bf16*)alloc((long long)B_ * N_ * DKS, 2);
  __bf16* kSt    = (__bf16*)alloc((long long)B_ * DKS * N_, 2);  // swizzled
  __bf16* attnSt = (__bf16*)alloc((long long)B_ * N_ * N_, 2);   // swizzled
  __bf16* out2   = (__bf16*)alloc((long long)B_ * T_ * N_, 2);

  auto cvt = [&](const float* s, __bf16* d, long long n) {
    int blocks = (int)((n + 255) / 256);
    if (blocks > 4096) blocks = 4096;
    f32_to_bf16_kernel<<<blocks, 256, 0, stream>>>(s, d, n);
  };
  auto cvtswz = [&](const float* s, __bf16* d, int rows, int cols) {
    long long n = (long long)rows * cols;
    int blocks = (int)((n + 255) / 256);
    if (blocks > 4096) blocks = 4096;
    f32_to_bf16_swz_kernel<<<blocks, 256, 0, stream>>>(s, d, rows, cols);
  };
  auto gemm = [&](const __bf16* A, long long sA, int lda,
                  const __bf16* Bm, long long sB, int ldb,
                  float* Cf, __bf16* Cb, long long sC, int ldc,
                  const float* bias, float scale,
                  int M, int Ncols, int K, int mode) {
    dim3 grid(Ncols / BN, M / BM, B_);
    gemm_bf16_kernel<<<grid, GEMM_THREADS, 0, stream>>>(
        A, sA, lda, Bm, sB, ldb, Cf, Cb, sC, ldc, bias, scale, K, mode);
  };

  const float scaleT = 0.125f;                 // 64^-0.5
  const float scaleS = 0.08838834764831845f;   // 128^-0.5

  // 0) downconvert; weights go straight to fragment-major swizzled layout
  cvt(x_T, xT_bf, (long long)B_ * T_ * N_);
  cvt(x_S, xS_bf, (long long)B_ * N_ * T_);
  cvtswz(Wq_T, wqT, N_, DKT);
  cvtswz(Wk_T, wkT, N_, DKT);
  cvtswz(Wv_T, wvT, N_, N_);
  cvtswz(Wq_S, wqS, T_, DKS);
  cvtswz(Wk_S, wkS, T_, DKS);
  cvtswz(Wout_T, wout, N_, N_);

  // 1) temporal projections (Q: *scale+bias; K: transposed+swizzled; V: swz)
  gemm(xT_bf, (long long)T_ * N_, N_, wqT, 0, DKT,
       nullptr, qT, (long long)T_ * DKT, DKT, bias_T, scaleT, T_, DKT, N_, 3);
  gemm(xT_bf, (long long)T_ * N_, N_, wkT, 0, DKT,
       nullptr, kTt, (long long)DKT * T_, T_, nullptr, 1.0f, T_, DKT, N_, 2);
  gemm(xT_bf, (long long)T_ * N_, N_, wvT, 0, N_,
       nullptr, vT, (long long)T_ * N_, N_, nullptr, 1.0f, T_, N_, N_, 5);

  // 2) temporal logits -> softmax -> attn @ V
  gemm(qT, (long long)T_ * DKT, DKT, kTt, (long long)DKT * T_, T_,
       logits, nullptr, (long long)T_ * T_, T_, nullptr, 1.0f, T_, T_, DKT, 0);
  softmax_bf16_kernel<<<B_ * T_, 256, 0, stream>>>(
      logits, attnT, T_, T_, (long long)T_ * T_, T_, 0);
  gemm(attnT, (long long)T_ * T_, T_, vT, (long long)T_ * N_, N_,
       nullptr, outT, (long long)T_ * N_, N_, nullptr, 1.0f, T_, N_, T_, 1);

  // 3) spatial attention; softmax stores attn_S^T in fragment-major layout
  gemm(xS_bf, (long long)N_ * T_, T_, wqS, 0, DKS,
       nullptr, qS, (long long)N_ * DKS, DKS, bias_S, scaleS, N_, DKS, T_, 3);
  gemm(xS_bf, (long long)N_ * T_, T_, wkS, 0, DKS,
       nullptr, kSt, (long long)DKS * N_, N_, nullptr, 1.0f, N_, DKS, T_, 2);
  gemm(qS, (long long)N_ * DKS, DKS, kSt, (long long)DKS * N_, N_,
       logits, nullptr, (long long)N_ * N_, N_, nullptr, 1.0f, N_, N_, DKS, 0);
  softmax_bf16_kernel<<<B_ * N_, 256, 0, stream>>>(
      logits, attnSt, N_, N_, (long long)N_ * N_, N_, 1);

  // 4) cross mix, then output projection + bout, written transposed [b,n,t]
  gemm(outT, (long long)T_ * N_, N_, attnSt, (long long)N_ * N_, N_,
       nullptr, out2, (long long)T_ * N_, N_, nullptr, 1.0f, T_, N_, N_, 1);
  gemm(out2, (long long)T_ * N_, N_, wout, 0, N_,
       out, nullptr, (long long)N_ * T_, T_, bout_T, 1.0f, T_, N_, N_, 4);
}
